// GRIDGRU_19524921328053
// MI455X (gfx1250) — compile-verified, matrix operand order checked by reading, not run on hardware
//
#include <hip/hip_runtime.h>
#include <hip/hip_bf16.h>

// ---------------------------------------------------------------------------
// GridGRU forward for MI455X (gfx1250): wave32 WMMA bf16 GEMMs, TDM async
// tile staging (tensor_load_to_lds + s_wait_tensorcnt), persistent scan.
// ---------------------------------------------------------------------------

typedef __bf16 bf16;
typedef __attribute__((ext_vector_type(8)))  __bf16 bf16x8;
typedef __attribute__((ext_vector_type(16))) __bf16 bf16x16;
typedef __attribute__((ext_vector_type(8)))  float  f32x8;
typedef __attribute__((ext_vector_type(4)))  unsigned int u32x4;
typedef __attribute__((ext_vector_type(8)))  int i32x8;
typedef __attribute__((ext_vector_type(4)))  int i32x4;

#define N_   32
#define T_   512
#define D_   1024
#define H_   1024
#define DH_  2048          // D+H
#define C3_  3072          // 3*H
#define NT_  16384         // N*T
#define WROWS 2048
#define WCOLS 6144

#define BM 128             // GEMM block tile M
#define BN 256             // GEMM block tile N
#define BK 32
#define LDSS 40            // padded LDS row stride: 32 data + 8 pad bf16 (80B rows)
#define SCAN_BLOCKS 16

__device__ __forceinline__ float sigmoidf_(float x) { return 1.0f / (1.0f + __expf(-x)); }

__device__ __forceinline__ f32x8 wmma_bf16(bf16x16 a, bf16x16 b, f32x8 c) {
  return __builtin_amdgcn_wmma_f32_16x16x32_bf16(false, a, false, b, (short)0, c, false, false);
}

// Per-lane fragment gather for 16-bit 16x32 A (or 32x16 B stored [n][k]):
// lane L holds row (L&15); halves at K = [8*kh, +8) and [16+8*kh, +8).
__device__ __forceinline__ bf16x16 frag_ld(const bf16* base, int kh) {
  bf16x8 lo = *(const bf16x8*)(base + 8 * kh);
  bf16x8 hi = *(const bf16x8*)(base + 16 + 8 * kh);
  return __builtin_shufflevector(lo, hi, 0,1,2,3,4,5,6,7,8,9,10,11,12,13,14,15);
}

// ---------------------------------------------------------------------------
// TDM: async DMA of a 2D bf16 tile (tile_rows x 32) global -> LDS.
// D# per CDNA5 ISA: data_size=2B, pad_enable with
// pad_interval=3 (16 dwords = one 32-elem row) and pad_amount=3 (4 dwords =
// 8 bf16) reproduces the LDSS=40 padded layout in LDS.
// ---------------------------------------------------------------------------
__device__ __forceinline__ void tdm_load_tile(unsigned lds_off, const bf16* gsrc,
                                              unsigned row_stride, int tile_rows) {
  unsigned long long ga = (unsigned long long)(size_t)gsrc;
  u32x4 g0;
  g0[0] = 1u;                                          // count=1, user descriptor
  g0[1] = lds_off;                                     // LDS byte address
  g0[2] = (unsigned)ga;                                // global_addr lo
  g0[3] = (unsigned)(ga >> 32) | (2u << 30);           // global_addr hi | type=2
  unsigned td0 = row_stride;                           // tensor_dim0 (elements)
  unsigned td1 = 0x00100000u;                          // tensor_dim1 (large, OOB only)
  i32x8 g1;
  g1[0] = (int)((1u << 16) | (1u << 20) | (3u << 22) | (3u << 25));
  g1[1] = (int)((td0 & 0xFFFFu) << 16);                // tensor_dim0[15:0]
  g1[2] = (int)((td0 >> 16) | ((td1 & 0xFFFFu) << 16));
  g1[3] = (int)((td1 >> 16) | ((unsigned)BK << 16));   // tile_dim0 = 32
  g1[4] = tile_rows;                                   // tile_dim1 (tile_dim2 = 0)
  g1[5] = (int)td0;                                    // tensor_dim0_stride lo
  g1[6] = 0;
  g1[7] = 0;
  i32x4 z4 = {0, 0, 0, 0};
#if defined(__clang_major__) && (__clang_major__ >= 23)
  i32x8 z8 = {0, 0, 0, 0, 0, 0, 0, 0};
  __builtin_amdgcn_tensor_load_to_lds(g0, g1, z4, z4, z8, 0);
#else
  __builtin_amdgcn_tensor_load_to_lds(g0, g1, z4, z4, 0);
#endif
}

__device__ __forceinline__ unsigned lds_off_of(const void* p) {
  return (unsigned)(size_t)p;                          // LDS aperture: low 32 bits
}

// ---------------------------------------------------------------------------
// prep kernels
// ---------------------------------------------------------------------------
__global__ __launch_bounds__(256) void k_conv_w(const float* __restrict__ W,
                                                bf16* __restrict__ Wt) {
  size_t idx = (size_t)blockIdx.x * 256 + threadIdx.x;       // over WROWS*WCOLS
  size_t k = idx / WCOLS, n = idx % WCOLS;
  Wt[n * WROWS + k] = (bf16)W[idx];                          // Wt[n][k] = W[k][n]
}

__global__ __launch_bounds__(256) void k_conv_x(const float* __restrict__ x,
                                                bf16* __restrict__ xh) {
  size_t idx = (size_t)blockIdx.x * 256 + threadIdx.x;       // over NT_*D_
  size_t row = idx >> 10, c = idx & 1023;
  xh[row * DH_ + c] = (bf16)x[idx];
}

__global__ __launch_bounds__(256) void k_init_h(const float* __restrict__ prev,
                                                float* __restrict__ h32,
                                                bf16* __restrict__ h16) {
  int idx = blockIdx.x * 256 + threadIdx.x;                  // over N_*H_
  float v = prev[idx];
  h32[idx] = v;
  h16[idx] = (bf16)v;
}

__global__ void k_zero(unsigned* p) { if (threadIdx.x == 0) *p = 0u; }

// ---------------------------------------------------------------------------
// Main WMMA GEMM:  C[M x Ncols] = A[M x K] * B^T  (B given as Wt[n][k]).
// 256 threads = 8 waves (2x4), 64x64 wave tile = 4x4 wmma accumulators.
// Tiles staged by the Tensor Data Mover into double-buffered LDS; one
// s_wait_tensorcnt + barrier per K-chunk; DMA overlaps the WMMA stream.
// EPI: 0 = +bias store (gates)   1 = sigmoid -> ud / rd*x (grd)
//      2 = raw store (c1)        3 = tanh + final blend -> output h
// ---------------------------------------------------------------------------
template <int EPI>
__global__ __launch_bounds__(256) void k_gemm(
    const bf16* __restrict__ A, int lda,
    const bf16* __restrict__ B, int ldb,
    float* __restrict__ C, int ldc, int K,
    const float* __restrict__ bias,
    const float* __restrict__ e0,   // EPI3: c1
    const float* __restrict__ e1,   // EPI3: ud
    const float* __restrict__ e2,   // EPI1/3: x (f32, ld=D_)
    float* __restrict__ o0,         // EPI1: ud out
    bf16* __restrict__ o1)          // EPI1: rd*x out (bf16)
{
  __shared__ bf16 As[2][BM * LDSS];
  __shared__ bf16 Bs[2][BN * LDSS];

  const int tid  = threadIdx.x;
  const int lane = tid & 31, wave = tid >> 5;
  const int wm = wave >> 2, wn = wave & 3;         // 2 x 4 wave grid, 64x64 each
  const int m0 = blockIdx.x * BM, n0 = blockIdx.y * BN;
  const int r = lane & 15, kh = lane >> 4;

  f32x8 acc[4][4];
#pragma unroll
  for (int m = 0; m < 4; ++m)
#pragma unroll
    for (int n = 0; n < 4; ++n)
#pragma unroll
      for (int i = 0; i < 8; ++i) acc[m][n][i] = 0.0f;

  const bf16* Atile = A + (size_t)m0 * lda;        // 128 rows, stride lda
  const bf16* Btile = B + (size_t)n0 * ldb;        // 256 rows, stride ldb

  // prologue: TDM chunk 0 into buffer 0
  if (wave == 0) {
    tdm_load_tile(lds_off_of(&As[0][0]), Atile, (unsigned)lda, BM);
    tdm_load_tile(lds_off_of(&Bs[0][0]), Btile, (unsigned)ldb, BN);
    __builtin_amdgcn_s_wait_tensorcnt(0);
  }
  __syncthreads();

  int cur = 0;
  for (int k0 = 0; k0 < K; k0 += BK) {
    const bool more = (k0 + BK) < K;
    if (more && wave == 0) {                       // async prefetch next chunk
      tdm_load_tile(lds_off_of(&As[cur ^ 1][0]), Atile + (k0 + BK), (unsigned)lda, BM);
      tdm_load_tile(lds_off_of(&Bs[cur ^ 1][0]), Btile + (k0 + BK), (unsigned)ldb, BN);
    }

    bf16x16 bfv[4];
#pragma unroll
    for (int n = 0; n < 4; ++n)
      bfv[n] = frag_ld(&Bs[cur][(wn * 64 + 16 * n + r) * LDSS], kh);
#pragma unroll
    for (int m = 0; m < 4; ++m) {
      bf16x16 av = frag_ld(&As[cur][(wm * 64 + 16 * m + r) * LDSS], kh);
#pragma unroll
      for (int n = 0; n < 4; ++n)
        acc[m][n] = wmma_bf16(av, bfv[n], acc[m][n]);
    }
    if (more) {
      if (wave == 0) __builtin_amdgcn_s_wait_tensorcnt(0);
      __syncthreads();                             // one barrier per chunk
      cur ^= 1;
    }
  }

  // epilogue: C layout — VGPR i: M = i + 8*(lane>>4), N = lane&15
#pragma unroll
  for (int m = 0; m < 4; ++m)
#pragma unroll
    for (int n = 0; n < 4; ++n)
#pragma unroll
      for (int i = 0; i < 8; ++i) {
        size_t row = (size_t)(m0 + wm * 64 + 16 * m + i + 8 * kh);
        int    col = n0 + wn * 64 + 16 * n + r;
        float  v   = acc[m][n][i];
        if constexpr (EPI == 0) {                       // gates = acc + bt
          C[row * ldc + col] = v + bias[col];
        } else if constexpr (EPI == 1) {                // grd = sigmoid(bd + acc)
          float g = sigmoidf_(v + bias[col]);
          if (col < D_) o0[row * D_ + col] = g;         // ud
          else o1[row * D_ + (col - D_)] =              // rd * x (bf16 for next GEMM)
              (bf16)(g * e2[row * D_ + (col - D_)]);
        } else if constexpr (EPI == 2) {                // c1 raw
          C[row * ldc + col] = v;
        } else {                                        // EPI 3: final blend
          float hcd = tanhf(v + bias[col] + e0[row * D_ + col]);
          float xx  = e2[row * D_ + col];
          C[row * ldc + col] = xx + e1[row * D_ + col] * (hcd - xx);
        }
      }
}

// ---------------------------------------------------------------------------
// Persistent recurrent scan: 16 blocks resident, device-wide barrier between
// the gate phase and the candidate phase of every step. Monotonic counter.
// ---------------------------------------------------------------------------
__device__ __forceinline__ void gbar(unsigned* ctr, int tid) {
  __syncthreads();
  if (tid == 0) {
    __threadfence();                                 // release prior writes
    unsigned my = __hip_atomic_fetch_add(ctr, 1u, __ATOMIC_ACQ_REL,
                                         __HIP_MEMORY_SCOPE_AGENT) + 1u;
    unsigned need = ((my + SCAN_BLOCKS - 1u) / SCAN_BLOCKS) * SCAN_BLOCKS;
    while (__hip_atomic_load(ctr, __ATOMIC_ACQUIRE, __HIP_MEMORY_SCOPE_AGENT) < need)
      __builtin_amdgcn_s_sleep(1);
  }
  __syncthreads();
  __threadfence();                                   // acquire for all threads
}

// One 128-col tile of a 32-row step GEMM (K = 1024), 8 waves = 8 col strips.
// B tile (128x32) streamed by TDM into double-buffered LDS; A tile (32x32)
// register-staged by the first two waves.
// PH 0: gg = sigmoid(gates_t + h@Whtg) -> u_buf, rh = r*h
// PH 1: hc = tanh(gates_t + rh@Whtc); hn = h + u*(hc-h) -> h32/h16 + xh slot
template <int PH>
__device__ void scan_tile(int n0,
                          const bf16* __restrict__ Amat,   // 32x1024, lda=1024
                          const bf16* __restrict__ Bmat,   // Wt + offset, ldb=2048
                          const float* __restrict__ gt,    // gates + t*3072
                          float* __restrict__ h32,
                          float* __restrict__ u_buf,
                          bf16* __restrict__ rh_out,
                          bf16* __restrict__ h16_out,
                          bf16* __restrict__ xh,
                          int t, bf16 (*As)[32 * LDSS], bf16 (*Bs)[128 * LDSS]) {
  const int tid  = threadIdx.x;
  const int lane = tid & 31, wave = tid >> 5;
  const int r = lane & 15, kh = lane >> 4;

  f32x8 acc[2];
#pragma unroll
  for (int m = 0; m < 2; ++m)
#pragma unroll
    for (int i = 0; i < 8; ++i) acc[m][i] = 0.0f;

  const bf16* Btile = Bmat + (size_t)n0 * DH_;
  const int lrow = tid >> 1, lh = tid & 1;

  // prologue: chunk 0
  if (wave == 0) tdm_load_tile(lds_off_of(&Bs[0][0]), Btile, DH_, 128);
  if (tid < 64) {
    const bf16x8* ga = (const bf16x8*)(Amat + (size_t)lrow * H_ + 16 * lh);
    bf16x8 a0 = ga[0], a1 = ga[1];
    *(bf16x8*)&As[0][lrow * LDSS + 16 * lh]     = a0;
    *(bf16x8*)&As[0][lrow * LDSS + 16 * lh + 8] = a1;
  }
  if (wave == 0) __builtin_amdgcn_s_wait_tensorcnt(0);
  __syncthreads();

  int cur = 0;
  for (int k0 = 0; k0 < H_; k0 += BK) {
    const bool more = (k0 + BK) < H_;
    bf16x8 a0, a1;
    if (more) {
      if (wave == 0)
        tdm_load_tile(lds_off_of(&Bs[cur ^ 1][0]), Btile + (k0 + BK), DH_, 128);
      if (tid < 64) {
        const bf16x8* ga = (const bf16x8*)(Amat + (size_t)lrow * H_ + k0 + BK + 16 * lh);
        a0 = ga[0]; a1 = ga[1];
      }
    }

    bf16x16 bv = frag_ld(&Bs[cur][(wave * 16 + r) * LDSS], kh);
#pragma unroll
    for (int m = 0; m < 2; ++m) {
      bf16x16 av = frag_ld(&As[cur][(16 * m + r) * LDSS], kh);
      acc[m] = wmma_bf16(av, bv, acc[m]);
    }

    if (more) {
      if (tid < 64) {
        *(bf16x8*)&As[cur ^ 1][lrow * LDSS + 16 * lh]     = a0;
        *(bf16x8*)&As[cur ^ 1][lrow * LDSS + 16 * lh + 8] = a1;
      }
      if (wave == 0) __builtin_amdgcn_s_wait_tensorcnt(0);
      __syncthreads();
      cur ^= 1;
    }
  }

  const size_t grs = (size_t)T_ * C3_;               // gates row stride (row = batch)
#pragma unroll
  for (int m = 0; m < 2; ++m)
#pragma unroll
    for (int i = 0; i < 8; ++i) {
      int row = 16 * m + i + 8 * kh;                 // batch index 0..31
      int col = n0 + wave * 16 + r;
      float v = acc[m][i];
      if constexpr (PH == 0) {
        float g = sigmoidf_(gt[(size_t)row * grs + col] + v);
        if (col < H_) u_buf[row * H_ + col] = g;
        else rh_out[row * H_ + (col - H_)] =
            (bf16)(g * h32[row * H_ + (col - H_)]);
      } else {
        float hc = tanhf(gt[(size_t)row * grs + 2 * H_ + col] + v);
        float ho = h32[row * H_ + col];
        float hn = ho + u_buf[row * H_ + col] * (hc - ho);
        h32[row * H_ + col] = hn;
        h16_out[row * H_ + col] = (bf16)hn;
        xh[((size_t)row * T_ + t) * DH_ + H_ + col] = (bf16)hn;
      }
    }
}

__global__ __launch_bounds__(256) void k_scan_persist(
    const bf16* __restrict__ Wt,
    const float* __restrict__ gates,
    float* __restrict__ h32,
    bf16* __restrict__ h16,
    float* __restrict__ u_buf,
    bf16* __restrict__ rh,
    bf16* __restrict__ xh,
    float* __restrict__ out_last,       // d_out + NT_*D_
    unsigned* __restrict__ ctr)
{
  __shared__ bf16 As[2][32 * LDSS];
  __shared__ bf16 Bs[2][128 * LDSS];
  const int blk = blockIdx.x, tid = threadIdx.x;
  const bf16* Bg = Wt + 1024;                           // Whtg^T (rows 0..2047, k off 1024)
  const bf16* Bc = Wt + (size_t)2048 * WROWS + 1024;    // Whtc^T

  for (int t = 0; t < T_; ++t) {
    const float* gt = gates + (size_t)t * C3_;
    scan_tile<0>(blk * 128, h16, Bg, gt, h32, u_buf, rh, nullptr, nullptr, t, As, Bs);
    gbar(ctr, tid);
    if (blk < 8)
      scan_tile<1>(blk * 128, rh, Bc, gt, h32, u_buf, nullptr, h16, xh, t, As, Bs);
    gbar(ctr, tid);
  }
  for (int idx = blk * 256 + tid; idx < N_ * H_; idx += SCAN_BLOCKS * 256)
    out_last[idx] = h32[idx];
}

// ---------------------------------------------------------------------------
// host side
// ---------------------------------------------------------------------------
extern "C" void kernel_launch(void* const* d_in, const int* in_sizes, int n_in,
                              void* d_out, int out_size, void* d_ws, size_t ws_size,
                              hipStream_t stream) {
  const float* x      = (const float*)d_in[0];
  const float* prevht = (const float*)d_in[1];
  const float* weight = (const float*)d_in[2];
  const float* bias   = (const float*)d_in[3];
  float* out = (float*)d_out;

  char* ws = (char*)d_ws;
  size_t off = 0;
  auto take = [&](size_t bytes) {
    off = (off + 255) & ~(size_t)255; void* p = ws + off; off += bytes; return p;
  };

  bf16*  Wt    = (bf16*) take((size_t)WCOLS * WROWS * 2);     // transposed bf16 weight
  bf16*  xh    = (bf16*) take((size_t)NT_ * DH_ * 2);         // [x | ht] bf16 concat
  float* gates = (float*)take((size_t)NT_ * C3_ * 4);
  float* c1    = (float*)take((size_t)NT_ * D_ * 4);
  float* ud    = (float*)take((size_t)NT_ * D_ * 4);
  bf16*  rdx   = (bf16*) take((size_t)NT_ * D_ * 2);
  float* h32   = (float*)take((size_t)N_ * H_ * 4);
  bf16*  h16   = (bf16*) take((size_t)N_ * H_ * 2);
  float* u_buf = (float*)take((size_t)N_ * H_ * 4);
  bf16*  rh    = (bf16*) take((size_t)N_ * H_ * 2);
  unsigned* ctr = (unsigned*)take(64);

  // prep
  k_conv_w<<<(WROWS * WCOLS) / 256, 256, 0, stream>>>(weight, Wt);
  k_conv_x<<<((size_t)NT_ * D_) / 256, 256, 0, stream>>>(x, xh);
  k_init_h<<<(N_ * H_) / 256, 256, 0, stream>>>(prevht, h32, h16);
  k_zero<<<1, 32, 0, stream>>>(ctr);

  // gates = x @ Wxt + bt          (16384x1024)@(1024x3072)
  k_gemm<0><<<dim3(NT_ / BM, C3_ / BN), 256, 0, stream>>>(
      xh, DH_, Wt, WROWS, gates, C3_, D_, bias,
      nullptr, nullptr, nullptr, nullptr, nullptr);

  // recurrent scan, one persistent kernel for all 512 steps (+ last_ht copy)
  k_scan_persist<<<SCAN_BLOCKS, 256, 0, stream>>>(
      Wt, gates, h32, h16, u_buf, rh, xh, out + (size_t)NT_ * D_, ctr);

  // grd = sigmoid(bd + [x|ht] @ W[:,3H:3H+2D])   K = 2048, one GEMM
  k_gemm<1><<<dim3(NT_ / BM, (2 * D_) / BN), 256, 0, stream>>>(
      xh, DH_, Wt + (size_t)3072 * WROWS, WROWS, nullptr, 0, DH_, bias + C3_,
      nullptr, nullptr, x, ud, rdx);

  // c1 = ht @ Whd[:, 2D:]         K = 1024, raw store
  k_gemm<2><<<dim3(NT_ / BM, D_ / BN), 256, 0, stream>>>(
      xh + H_, DH_, Wt + (size_t)5120 * WROWS + 1024, WROWS, c1, D_, D_,
      nullptr, nullptr, nullptr, nullptr, nullptr, nullptr);

  // hcd = tanh(bd3 + c1 + (rd*x) @ Wxd[:, 2D:]);  out = x + ud*(hcd - x)
  k_gemm<3><<<dim3(NT_ / BM, D_ / BN), 256, 0, stream>>>(
      rdx, D_, Wt + (size_t)5120 * WROWS, WROWS, out, D_, D_,
      bias + C3_ + 2 * D_, c1, ud, x, nullptr, nullptr);
}
